// CasualGroupQueryAttention_74053826117889
// MI455X (gfx1250) — compile-verified, matrix operand order checked by reading, not run on hardware
//
#include <hip/hip_runtime.h>
#include <hip/hip_bf16.h>

// ---------------- problem constants ----------------
#define BATCH 2
#define TLEN  2048
#define DMODEL 2048
#define NHEAD 16
#define NGROUP 4
#define HSZ   128
#define KVDIM 1024          // 2 * NGROUP * HSZ
#define KD    512           // NGROUP * HSZ
#define MROWS (BATCH*TLEN)  // 4096

// ---------------- WMMA types ----------------
typedef __attribute__((ext_vector_type(16))) __bf16        v16bf;
typedef __attribute__((ext_vector_type(8)))  float         v8f;
typedef __attribute__((ext_vector_type(4)))  unsigned int  u32x4;

union FragU { v16bf v; u32x4 u[2]; };

__device__ __forceinline__ unsigned short f2bf(float f) {
    unsigned int u = __float_as_uint(f);
    unsigned int r = (u + 0x7fffu + ((u >> 16) & 1u)) >> 16;
    return (unsigned short)r;
}
__device__ __forceinline__ float bf2f(unsigned short h) {
    return __uint_as_float(((unsigned int)h) << 16);
}

// Load a 16x32 bf16 A/B fragment (both operands of an A*B^T GEMM use the
// identical pattern): lane L -> row L%16, two b128 chunks at k=(L/16)*8 and +16.
__device__ __forceinline__ v16bf load_frag(const unsigned short* base,
                                           size_t rowpitch, int lane) {
    const int m  = lane & 15;
    const int kb = (lane >> 4) << 3;
    const unsigned short* p = base + (size_t)m * rowpitch + kb;
    FragU f;
    f.u[0] = *(const u32x4*)(p);
    f.u[1] = *(const u32x4*)(p + 16);
    return f.v;
}

// Prefetch the (per-lane) cacheline of a fragment row `ahead` k-elements ahead.
// Lowers to global_prefetch_b8 (WGP scope: pulls through L2 into near caches).
__device__ __forceinline__ void prefetch_frag(const unsigned short* base,
                                              size_t rowpitch, int lane) {
    const int m  = lane & 15;
    const int kb = (lane >> 4) << 3;
    __builtin_prefetch(base + (size_t)m * rowpitch + kb, 0, 3);
}

__device__ __forceinline__ v8f wmma_bf16(v16bf a, v16bf b, v8f c) {
    return __builtin_amdgcn_wmma_f32_16x16x32_bf16(
        false, a, false, b, (short)0, c, false, false);
}

// ---------------- cast f32 -> bf16 ----------------
__global__ __launch_bounds__(256) void cast_bf16_kernel(
    const float* __restrict__ in, unsigned short* __restrict__ out, size_t n) {
    size_t i = (size_t)blockIdx.x * blockDim.x + threadIdx.x;
    size_t stride = (size_t)gridDim.x * blockDim.x;
    for (; i < n; i += stride) out[i] = f2bf(in[i]);
}

// ---------------- generic WMMA GEMM: C[m,n] = sum_k A[m,k]*W[n,k] (+bias) ----
// MODE 0: f32 store (output projection)   MODE 1: bf16 store + bias (Q proj)
// MODE 2: KV projection: n<512 -> K row-major; n>=512 -> V transposed
template <int MODE>
__global__ __launch_bounds__(32) void gemm_bf16_kernel(
    const unsigned short* __restrict__ A,   // M x K
    const unsigned short* __restrict__ W,   // N x K
    const float* __restrict__ bias,
    float* __restrict__ Cf,
    unsigned short* __restrict__ Cb,
    unsigned short* __restrict__ Kb,
    unsigned short* __restrict__ Vtb,
    int M, int N, int K) {
    const int lane  = threadIdx.x & 31;
    const int laneh = lane >> 4;
    const int nlo   = lane & 15;
    const size_t tileN = (size_t)blockIdx.x * 64;
    const size_t tileM = (size_t)blockIdx.y * 64;

    v8f acc[4][4] = {};
    for (int k0 = 0; k0 < K; k0 += 32) {
        // Prefetch 4 k-steps (256B) ahead to hide HBM->L2->L0 latency.
        const int kpf = k0 + 128;
        if (kpf < K) {
#pragma unroll
            for (int i = 0; i < 4; ++i)
                prefetch_frag(A + (tileM + i * 16) * (size_t)K + kpf, K, lane);
#pragma unroll
            for (int j = 0; j < 4; ++j)
                prefetch_frag(W + (tileN + j * 16) * (size_t)K + kpf, K, lane);
        }
        v16bf af[4], bfv[4];
#pragma unroll
        for (int i = 0; i < 4; ++i)
            af[i] = load_frag(A + (tileM + i * 16) * (size_t)K + k0, K, lane);
#pragma unroll
        for (int j = 0; j < 4; ++j)
            bfv[j] = load_frag(W + (tileN + j * 16) * (size_t)K + k0, K, lane);
#pragma unroll
        for (int i = 0; i < 4; ++i)
#pragma unroll
            for (int j = 0; j < 4; ++j)
                acc[i][j] = wmma_bf16(af[i], bfv[j], acc[i][j]);
    }

#pragma unroll
    for (int i = 0; i < 4; ++i) {
#pragma unroll
        for (int j = 0; j < 4; ++j) {
#pragma unroll
            for (int r = 0; r < 8; ++r) {
                const size_t row = tileM + i * 16 + r + laneh * 8;
                const size_t col = tileN + j * 16 + nlo;
                float v = acc[i][j][r];
                if (MODE == 0) {
                    Cf[row * (size_t)N + col] = v;
                } else if (MODE == 1) {
                    Cb[row * (size_t)N + col] = f2bf(v + bias[col]);
                } else {
                    v += bias[col];
                    if (col < KD) {
                        Kb[row * (size_t)KD + col] = f2bf(v);
                    } else {
                        const int nv = (int)col - KD;
                        const int g = nv >> 7, dd = nv & 127;
                        const size_t bb = row / TLEN, tt = row % TLEN;
                        Vtb[(((bb * NGROUP + g) * (size_t)HSZ) + dd) * TLEN + tt] =
                            f2bf(v);
                    }
                }
            }
        }
    }
}

// ---------------- RoPE (in-place on bf16 Q and K) ----------------
// slot 0..15 = Q heads, 16..19 = K groups; each thread does one (i, i+64) pair.
__global__ __launch_bounds__(256) void rope_kernel(
    unsigned short* __restrict__ qb, unsigned short* __restrict__ kb,
    const float* __restrict__ cosp, const float* __restrict__ sinp) {
    const size_t tid = (size_t)blockIdx.x * blockDim.x + threadIdx.x;
    const int i = (int)(tid & 63);
    const size_t hr = tid >> 6;
    const int slot = (int)(hr % 20);
    const size_t bt = hr / 20;
    if (bt >= (size_t)MROWS) return;
    const int t = (int)(bt % TLEN);
    unsigned short* base =
        (slot < NHEAD) ? (qb + bt * DMODEL + (size_t)slot * HSZ)
                       : (kb + bt * KD + (size_t)(slot - NHEAD) * HSZ);
    const float u1 = bf2f(base[i]);
    const float u2 = bf2f(base[i + 64]);
    const float c1 = cosp[(size_t)t * HSZ + i];
    const float s1 = sinp[(size_t)t * HSZ + i];
    const float c2 = cosp[(size_t)t * HSZ + 64 + i];
    const float s2 = sinp[(size_t)t * HSZ + 64 + i];
    base[i]      = f2bf(u1 * c1 - u2 * s1);
    base[i + 64] = f2bf(u2 * c2 + u1 * s2);
}

// ---------------- causal flash attention ----------------
// grid = (T/16, NHEAD, BATCH), 1 wave per block. Q tile (16x128) resident,
// 32-key tiles, online softmax, P via LDS -> A-fragment, O in 8 C-fragments.
__global__ __launch_bounds__(32) void flash_gqa_kernel(
    const unsigned short* __restrict__ qb,
    const unsigned short* __restrict__ kb,
    const unsigned short* __restrict__ vtb,
    unsigned short* __restrict__ ob) {
    __shared__ unsigned short plds[16][40];  // 16 x 32 P tile, padded pitch

    const int lane  = threadIdx.x & 31;
    const int laneh = lane >> 4;
    const int nlo   = lane & 15;
    const int qt = blockIdx.x, h = blockIdx.y, b = blockIdx.z;
    const int g = h >> 2;  // N_REP = 4
    const int qbase = qt * 16;

    const unsigned short* qp = qb + ((size_t)(b * TLEN + qbase)) * DMODEL + h * HSZ;
    const unsigned short* kp = kb + ((size_t)b * TLEN) * KD + g * HSZ;
    const unsigned short* vp = vtb + ((size_t)(b * NGROUP + g)) * HSZ * TLEN;

    v16bf Qf[4];
#pragma unroll
    for (int j = 0; j < 4; ++j) Qf[j] = load_frag(qp + j * 32, DMODEL, lane);

    v8f acc[8] = {};
    float mrun[8], lrun[8];
#pragma unroll
    for (int r = 0; r < 8; ++r) { mrun[r] = -3.0e38f; lrun[r] = 0.0f; }

    const float SCALE = 0.08838834764831845f;  // 1/sqrt(128)
    const float L2E   = 1.4426950408889634f;

    const int ktmax = (qbase + 15) >> 5;
    for (int kt = 0; kt <= ktmax; ++kt) {
        const int k0 = kt << 5;

        // Prefetch next key/value tiles while this one computes.
        if (kt < ktmax) {
            const int kn = k0 + 32;
#pragma unroll
            for (int j = 0; j < 4; ++j) {
                prefetch_frag(kp + (size_t)kn * KD + j * 32, KD, lane);
                prefetch_frag(kp + (size_t)(kn + 16) * KD + j * 32, KD, lane);
            }
#pragma unroll
            for (int d = 0; d < 8; ++d)
                prefetch_frag(vp + (size_t)(d * 16) * TLEN + kn, TLEN, lane);
        }

        v8f s0 = {}, s1 = {};
#pragma unroll
        for (int j = 0; j < 4; ++j) {
            v16bf kf0 = load_frag(kp + (size_t)k0 * KD + j * 32, KD, lane);
            v16bf kf1 = load_frag(kp + (size_t)(k0 + 16) * KD + j * 32, KD, lane);
            s0 = wmma_bf16(Qf[j], kf0, s0);
            s1 = wmma_bf16(Qf[j], kf1, s1);
        }
        const int col0 = k0 + nlo;
        float oscale[8];
#pragma unroll
        for (int r = 0; r < 8; ++r) {
            const int row = qbase + r + laneh * 8;
            float a = s0[r] * SCALE; if (col0 > row)      a = -3.0e38f;
            float c = s1[r] * SCALE; if (col0 + 16 > row) c = -3.0e38f;
            float mx = fmaxf(a, c);
            mx = fmaxf(mx, __shfl_xor(mx, 1, 32));
            mx = fmaxf(mx, __shfl_xor(mx, 2, 32));
            mx = fmaxf(mx, __shfl_xor(mx, 4, 32));
            mx = fmaxf(mx, __shfl_xor(mx, 8, 32));
            const float mnew = fmaxf(mrun[r], mx);
            const float sc = exp2f((mrun[r] - mnew) * L2E);
            const float p0 = exp2f((a - mnew) * L2E);
            const float p1 = exp2f((c - mnew) * L2E);
            float rs = p0 + p1;
            rs += __shfl_xor(rs, 1, 32);
            rs += __shfl_xor(rs, 2, 32);
            rs += __shfl_xor(rs, 4, 32);
            rs += __shfl_xor(rs, 8, 32);
            lrun[r] = lrun[r] * sc + rs;
            mrun[r] = mnew;
            oscale[r] = sc;
            plds[r + laneh * 8][nlo]      = f2bf(p0);
            plds[r + laneh * 8][16 + nlo] = f2bf(p1);
        }
#pragma unroll
        for (int d = 0; d < 8; ++d)
#pragma unroll
            for (int r = 0; r < 8; ++r) acc[d][r] *= oscale[r];
        __syncthreads();
        const v16bf pf = load_frag(&plds[0][0], 40, lane);
#pragma unroll
        for (int d = 0; d < 8; ++d) {
            v16bf vf = load_frag(vp + (size_t)(d * 16) * TLEN + k0, TLEN, lane);
            acc[d] = wmma_bf16(pf, vf, acc[d]);
        }
        __syncthreads();
    }

    unsigned short* op = ob + ((size_t)(b * TLEN + qbase)) * DMODEL + h * HSZ;
#pragma unroll
    for (int d = 0; d < 8; ++d)
#pragma unroll
        for (int r = 0; r < 8; ++r)
            op[(size_t)(r + laneh * 8) * DMODEL + d * 16 + nlo] =
                f2bf(acc[d][r] / lrun[r]);
}

// ---------------- host launcher ----------------
extern "C" void kernel_launch(void* const* d_in, const int* in_sizes, int n_in,
                              void* d_out, int out_size, void* d_ws, size_t ws_size,
                              hipStream_t stream) {
    const float* x    = (const float*)d_in[0];
    const float* cosp = (const float*)d_in[1];
    const float* sinp = (const float*)d_in[2];
    const float* Wq   = (const float*)d_in[3];
    const float* bq   = (const float*)d_in[4];
    const float* Wkv  = (const float*)d_in[5];
    const float* bkv  = (const float*)d_in[6];
    const float* Wo   = (const float*)d_in[7];
    float* out = (float*)d_out;

    char* ws = (char*)d_ws;
    size_t off = 0;
    auto alloc = [&](size_t elems) {
        unsigned short* p = (unsigned short*)(ws + off);
        off = (off + elems * 2 + 255) & ~(size_t)255;
        return p;
    };
    unsigned short* xb   = alloc((size_t)MROWS * DMODEL);
    unsigned short* wqb  = alloc((size_t)DMODEL * DMODEL);
    unsigned short* wkvb = alloc((size_t)KVDIM * DMODEL);
    unsigned short* wob  = alloc((size_t)DMODEL * DMODEL);
    unsigned short* qbuf = alloc((size_t)MROWS * DMODEL);
    unsigned short* kbuf = alloc((size_t)MROWS * KD);
    unsigned short* vtb  = alloc((size_t)BATCH * NGROUP * HSZ * TLEN);
    unsigned short* obuf = alloc((size_t)MROWS * DMODEL);

    auto cast = [&](const float* src, unsigned short* dst, size_t n) {
        int grid = (int)((n + 255) / 256);
        cast_bf16_kernel<<<grid, 256, 0, stream>>>(src, dst, n);
    };
    cast(x,   xb,   (size_t)MROWS * DMODEL);
    cast(Wq,  wqb,  (size_t)DMODEL * DMODEL);
    cast(Wkv, wkvb, (size_t)KVDIM * DMODEL);
    cast(Wo,  wob,  (size_t)DMODEL * DMODEL);

    // Q projection: (MROWS x D) x (D x D)^T + bq -> bf16
    gemm_bf16_kernel<1><<<dim3(DMODEL / 64, MROWS / 64), 32, 0, stream>>>(
        xb, wqb, bq, nullptr, qbuf, nullptr, nullptr, MROWS, DMODEL, DMODEL);

    // KV projection: (MROWS x D) x (KVDIM x D)^T + bkv -> K rowmajor + V^T
    gemm_bf16_kernel<2><<<dim3(KVDIM / 64, MROWS / 64), 32, 0, stream>>>(
        xb, wkvb, bkv, nullptr, nullptr, kbuf, vtb, MROWS, KVDIM, DMODEL);

    // RoPE on Q and K
    {
        size_t total = (size_t)MROWS * 20 * 64;
        rope_kernel<<<(int)(total / 256), 256, 0, stream>>>(qbuf, kbuf, cosp, sinp);
    }

    // Flash attention
    flash_gqa_kernel<<<dim3(TLEN / 16, NHEAD, BATCH), 32, 0, stream>>>(
        qbuf, kbuf, vtb, obuf);

    // Output projection: f32 result
    gemm_bf16_kernel<0><<<dim3(DMODEL / 64, MROWS / 64), 32, 0, stream>>>(
        obuf, wob, nullptr, out, nullptr, nullptr, nullptr, MROWS, DMODEL, DMODEL);
}
